// Model_59992103190876
// MI455X (gfx1250) — matrix operand order repaired and checked by reading, not compiled, not run on hardware
//
#include <hip/hip_runtime.h>
#include <hip/hip_bf16.h>

// ---------------------------------------------------------------------------
// MI455X (gfx1250, wave32) fused DSA-indexer prologue.
//   Q: RMSNorm*smooth -> bf16 WMMA GEMM (8192x2048 @ 2048x4096) -> RoPE (regs)
//      -> Hadamard WMMA -> absmax quant
//   K: LayerNorm -> bf16 WMMA GEMM (2048x128) -> RoPE -> Hadamard WMMA ->
//      quant -> scatter by k_cache_index
//   weights: x @ w_proj (fp32, shfl reductions, fused in prep)
// GEMM staging uses double-buffered global_load_async_to_lds_b128 (ASYNCcnt)
// when the toolchain exposes the builtin; otherwise a sync LDS copy.
// ---------------------------------------------------------------------------

typedef unsigned short u16;
typedef unsigned int   u32;
typedef float  v8f  __attribute__((ext_vector_type(8)));
typedef __bf16 v16bf __attribute__((ext_vector_type(16)));
typedef int    v4i  __attribute__((vector_size(16)));

union FragBF { v16bf v; u32 u[8]; };
union FragF  { v8f  v; float f[8]; };

#if defined(__has_builtin)
#if __has_builtin(__builtin_amdgcn_global_load_async_to_lds_b128)
#define HAVE_ASYNC 1
#endif
#endif
#ifndef HAVE_ASYNC
#define HAVE_ASYNC 0
#endif

#if HAVE_ASYNC
#if __has_builtin(__builtin_amdgcn_s_wait_asynccnt)
#define WAIT_ASYNC(n) __builtin_amdgcn_s_wait_asynccnt(n)
#else
#define WAIT_ASYNC(n) asm volatile("s_wait_asynccnt %0" ::"n"(n) : "memory")
#endif
#else
#define WAIT_ASYNC(n) ((void)0)
#endif

typedef __attribute__((address_space(1))) v4i* gv4i_p;  // global
typedef __attribute__((address_space(3))) v4i* lv4i_p;  // LDS

// copy 16 bytes global -> LDS (async DMA when available)
__device__ __forceinline__ void cp16(u16* lds, const u16* g) {
#if HAVE_ASYNC
  __builtin_amdgcn_global_load_async_to_lds_b128(
      (gv4i_p)(const void*)g, (lv4i_p)(void*)lds, 0, 0);
#else
  *(uint4*)lds = *(const uint4*)g;
#endif
}

__device__ __forceinline__ u16 f2bf(float f) {
  union { float f; u32 u; } c; c.f = f;
  u32 u = c.u;
  u += 0x7fffu + ((u >> 16) & 1u);   // round-to-nearest-even
  return (u16)(u >> 16);
}

// A fragment, 16x32 bf16: lanes 0-15 row=lane, K=0..15;
// lanes 16-31 row=lane-16, K=16..31. 8 dword LDS reads (merge to b128).
__device__ __forceinline__ v16bf load_a_frag(const u16* base, int ldr,
                                             int row, int kb, int kh) {
  FragBF f;
#pragma unroll
  for (int p = 0; p < 8; ++p) {
    int kbase = kh * 16 + 2 * p;
    f.u[p] = *(const u32*)(base + row * ldr + kb + kbase);
  }
  return f.v;
}

// B fragment, 32x16 bf16: lanes 0-15 hold K=0..15, lanes 16-31 K=16..31,
// N = lane&15. LDS holds B as [n][k] (weights pre-transposed in global).
__device__ __forceinline__ v16bf load_b_frag(const u16* base, int ldr,
                                             int n, int kb, int kh) {
  FragBF f;
#pragma unroll
  for (int p = 0; p < 8; ++p)
    f.u[p] = *(const u32*)(base + n * ldr + kb + kh * 16 + 2 * p);
  return f.v;
}

// ---------------------------------------------------------------------------
// prep: per-row stats + normalized bf16 activations + indexer weights
// ---------------------------------------------------------------------------
__global__ __launch_bounds__(256) void prep_kernel(
    const float* __restrict__ x, const float* __restrict__ q_norm,
    const float* __restrict__ q_norm_scale, const float* __restrict__ ln_g,
    const float* __restrict__ ln_b, const float* __restrict__ w_proj,
    u16* __restrict__ xq_bf, u16* __restrict__ xk_bf,
    float* __restrict__ weights) {
  __shared__ float xrow[2048];
  __shared__ float red[512];
  __shared__ float wsum[8 * 32];
  const int row = blockIdx.x;
  const int tid = threadIdx.x;
  const float* xb = x + (size_t)row * 2048;

  float s1 = 0.f, s2 = 0.f;
  float wacc[32];
#pragma unroll
  for (int e = 0; e < 32; ++e) wacc[e] = 0.f;

  for (int j = tid; j < 2048; j += 256) {
    float v = xb[j];
    xrow[j] = v;
    s1 += v;
    s2 += v * v;
    const float* wp = w_proj + (size_t)j * 32;
#pragma unroll
    for (int e = 0; e < 32; ++e) wacc[e] += v * wp[e];
  }
  red[tid] = s1;
  red[256 + tid] = s2;
  __syncthreads();
  for (int st = 128; st > 0; st >>= 1) {
    if (tid < st) {
      red[tid] += red[tid + st];
      red[256 + tid] += red[256 + tid + st];
    }
    __syncthreads();
  }
  const float mean = red[0] * (1.0f / 2048.0f);
  const float msq  = red[256] * (1.0f / 2048.0f);
  const float rms  = rsqrtf(msq + 1e-6f);                 // Q: rsqrt(E[x^2]+eps)
  const float inv  = rsqrtf(msq - mean * mean + 1e-6f);   // K: rsqrt(var+eps)

#pragma unroll
  for (int off = 16; off >= 1; off >>= 1)
#pragma unroll
    for (int e = 0; e < 32; ++e) wacc[e] += __shfl_xor(wacc[e], off, 32);
  const int wv = tid >> 5, lnn = tid & 31;
  wsum[wv * 32 + lnn] = wacc[lnn];
  __syncthreads();
  if (tid < 32) {
    float s = 0.f;
#pragma unroll
    for (int w = 0; w < 8; ++w) s += wsum[w * 32 + tid];
    weights[(size_t)row * 32 + tid] = s;
  }

  for (int j = tid; j < 2048; j += 256) {
    float v = xrow[j];
    xq_bf[(size_t)row * 2048 + j] = f2bf(v * rms * q_norm[j] * q_norm_scale[j]);
    xk_bf[(size_t)row * 2048 + j] = f2bf((v - mean) * inv * ln_g[j] + ln_b[j]);
  }
}

// ---------------------------------------------------------------------------
// fp32 -> bf16 (hadamards)
// ---------------------------------------------------------------------------
__global__ __launch_bounds__(256) void f2bf_kernel(const float* __restrict__ src,
                                                   u16* __restrict__ dst, int n) {
  int i = blockIdx.x * 256 + threadIdx.x;
  if (i < n) dst[i] = f2bf(src[i]);
}

// ---------------------------------------------------------------------------
// fp32 [K][N] -> bf16 [N][K] tiled transpose (one-time weight prep so the
// GEMM B staging becomes a contiguous async-DMA copy)
// ---------------------------------------------------------------------------
__global__ __launch_bounds__(256) void transpose_f2bf_kernel(
    const float* __restrict__ src, u16* __restrict__ dst, int K, int N) {
  __shared__ u16 tile[32][33];
  const int n0 = blockIdx.x * 32, k0 = blockIdx.y * 32;
  const int tx = threadIdx.x & 31, ty = threadIdx.x >> 5;
  for (int r = ty; r < 32; r += 8)
    tile[r][tx] = f2bf(src[(size_t)(k0 + r) * N + n0 + tx]);
  __syncthreads();
  for (int r = ty; r < 32; r += 8)
    dst[(size_t)(n0 + r) * K + k0 + tx] = tile[tx][r];
}

// ---------------------------------------------------------------------------
// Fused: GEMM1 (K=2048, bf16 WMMA, double-buffered async LDS staging) ->
// col scale -> RoPE (regs) -> GEMM2 (Hadamard, K=128) -> absmax quant ->
// store/scatter. Tile: 128 rows x 128 cols (one head); 8 wave32s.
// ---------------------------------------------------------------------------
template <bool IS_K>
__global__ __launch_bounds__(256) void fused_branch_kernel(
    const u16* __restrict__ a_bf,      // [8192][2048] bf16 activations
    const u16* __restrict__ bT_bf,     // [Ntot][2048] bf16 weights (transposed)
    const u16* __restrict__ had_bf,    // [128][128]   bf16 hadamard (k-major)
    const float* __restrict__ col_scale,  // w_qb_scale (Q) or nullptr
    const float* __restrict__ cosr,    // [4096][128]
    const float* __restrict__ sinr,    // [4096][128]
    const int* __restrict__ kidx,      // [8192] (K) or nullptr
    float* __restrict__ out_v,         // q_hif8 or k_hif8
    float* __restrict__ out_s) {       // q_scale or k_scale
  // Padded LDS strides (u16): 40 -> 80B rows, 136 -> 272B rows: conflict-free
  // 16-row fragment reads, 16B-aligned b128 staging.
  __shared__ u16 Abuf[2][128 * 40];
  __shared__ u16 Bbuf[2][128 * 40];
  __shared__ u16 ropeb[128 * 136];
  __shared__ u16 hadT[128 * 136];

  const int tid  = threadIdx.x;
  const int m0   = blockIdx.x * 128;
  const int head = blockIdx.y;
  const int n0   = head * 128;
  const int wave = tid >> 5, lane = tid & 31;
  const int ln = lane & 15, kh = lane >> 4;
  const int r0w = wave * 16;

  // stage hadamard transposed [n][k] once
  for (int i = tid; i < 128 * 128; i += 256) {
    int k = i >> 7, n = i & 127;
    hadT[n * 136 + k] = had_bf[i];
  }

  FragF acc[8];
#pragma unroll
  for (int t = 0; t < 8; ++t)
#pragma unroll
    for (int j = 0; j < 8; ++j) acc[t].f[j] = 0.f;

  // per-thread tile stage: 2x A + 2x B async b128 (4 ASYNCcnt ops/wave)
  auto stage = [&](int ib, int kb) {
#pragma unroll
    for (int h = 0; h < 2; ++h) {
      int l = tid + h * 256;
      int r = l >> 2, c8 = (l & 3) * 8;
      cp16(&Abuf[ib][r * 40 + c8],
           a_bf + (size_t)(m0 + r) * 2048 + kb + c8);
      cp16(&Bbuf[ib][r * 40 + c8],
           bT_bf + (size_t)(n0 + r) * 2048 + kb + c8);
    }
  };

  // ---------------- GEMM1: K = 2048, double-buffered ----------------
  stage(0, 0);
  int ib = 0;
  for (int kb = 0; kb < 2048; kb += 32, ib ^= 1) {
    if (kb + 32 < 2048) {
      stage(ib ^ 1, kb + 32);   // prefetch next tile into the other buffer
      WAIT_ASYNC(4);            // current tile done; next may stay in flight
    } else {
      WAIT_ASYNC(0);
    }
    __syncthreads();
    v16bf afrag = load_a_frag(Abuf[ib], 40, r0w + ln, 0, kh);
#pragma unroll
    for (int t = 0; t < 8; ++t) {
      v16bf bfrag = load_b_frag(Bbuf[ib], 40, t * 16 + ln, 0, kh);
      acc[t].v = __builtin_amdgcn_wmma_f32_16x16x32_bf16(
          false, afrag, false, bfrag, (short)0, acc[t].v, false, false);
    }
    __syncthreads();            // buffer free for the iteration after next
  }

  // ---------------- epilogue 1: column dequant scale (Q only) --------------
  if (!IS_K) {
#pragma unroll
    for (int t = 0; t < 8; ++t) {
      float sc = col_scale[n0 + t * 16 + ln];
#pragma unroll
      for (int j = 0; j < 8; ++j) acc[t].f[j] *= sc;
    }
  }

  // ---------------- RoPE in registers (d and d+64 are tiles t, t+4) --------
  int srow[8];
#pragma unroll
  for (int j = 0; j < 8; ++j)
    srow[j] = (m0 + r0w + (kh ? j + 8 : j)) & 4095;
#pragma unroll
  for (int t = 0; t < 4; ++t) {
    int d = t * 16 + ln;
#pragma unroll
    for (int j = 0; j < 8; ++j) {
      int s = srow[j];
      float c0 = cosr[s * 128 + d],      s0 = sinr[s * 128 + d];
      float c1 = cosr[s * 128 + d + 64], s1 = sinr[s * 128 + d + 64];
      float lo = acc[t].f[j], hi = acc[t + 4].f[j];
      acc[t].f[j]     = lo * c0 - hi * s0;
      acc[t + 4].f[j] = hi * c1 + lo * s1;
    }
  }

  // spill roped tile as bf16 for GEMM2 A fragments
#pragma unroll
  for (int t = 0; t < 8; ++t) {
    int d = t * 16 + ln;
#pragma unroll
    for (int j = 0; j < 8; ++j) {
      int r = r0w + (kh ? j + 8 : j);
      ropeb[r * 136 + d] = f2bf(acc[t].f[j]);
    }
  }
  __syncthreads();

  // ---------------- GEMM2: roped(128x128) @ hadamard(128x128) -------------
#pragma unroll
  for (int t = 0; t < 8; ++t)
#pragma unroll
    for (int j = 0; j < 8; ++j) acc[t].f[j] = 0.f;
  for (int kb = 0; kb < 128; kb += 32) {
    v16bf afrag = load_a_frag(ropeb, 136, r0w + ln, kb, kh);
#pragma unroll
    for (int t = 0; t < 8; ++t) {
      v16bf bfrag = load_b_frag(hadT, 136, t * 16 + ln, kb, kh);
      acc[t].v = __builtin_amdgcn_wmma_f32_16x16x32_bf16(
          false, afrag, false, bfrag, (short)0, acc[t].v, false, false);
    }
  }

  // ---------------- absmax quant: reduce across 16-lane half ---------------
  float am[8];
#pragma unroll
  for (int j = 0; j < 8; ++j) {
    am[j] = 0.f;
#pragma unroll
    for (int t = 0; t < 8; ++t) am[j] = fmaxf(am[j], fabsf(acc[t].f[j]));
  }
#pragma unroll
  for (int off = 8; off >= 1; off >>= 1)
#pragma unroll
    for (int j = 0; j < 8; ++j)
      am[j] = fmaxf(am[j], __shfl_xor(am[j], off, 32));

  float scale[8], inv[8];
  int grow[8], orow[8];
#pragma unroll
  for (int j = 0; j < 8; ++j) {
    scale[j] = fmaxf(am[j] * (1.0f / 127.0f), 1e-8f);
    inv[j] = 1.0f / scale[j];
    grow[j] = m0 + r0w + (kh ? j + 8 : j);
    orow[j] = IS_K ? kidx[grow[j]] : grow[j];
  }
#pragma unroll
  for (int t = 0; t < 8; ++t) {
    int col = t * 16 + ln;
#pragma unroll
    for (int j = 0; j < 8; ++j) {
      size_t o = IS_K ? ((size_t)orow[j] * 128 + col)
                      : ((size_t)orow[j] * 4096 + n0 + col);
      out_v[o] = rintf(acc[t].f[j] * inv[j]);
    }
  }
  if (ln == 0) {
#pragma unroll
    for (int j = 0; j < 8; ++j)
      out_s[IS_K ? (size_t)orow[j] : ((size_t)grow[j] * 32 + head)] = scale[j];
  }
}

// ---------------------------------------------------------------------------
extern "C" void kernel_launch(void* const* d_in, const int* in_sizes, int n_in,
                              void* d_out, int out_size, void* d_ws,
                              size_t ws_size, hipStream_t stream) {
  (void)in_sizes; (void)n_in; (void)out_size; (void)ws_size;
  const int H = 2048, Hn = 32, D = 128, L = 16384, BS = 8192;

  const float* x     = (const float*)d_in[0];
  const float* qn    = (const float*)d_in[1];
  const float* qns   = (const float*)d_in[2];
  const float* w_qb  = (const float*)d_in[3];
  const float* w_qbs = (const float*)d_in[4];
  const float* wk    = (const float*)d_in[5];
  const float* wproj = (const float*)d_in[6];
  const float* lng   = (const float*)d_in[7];
  const float* lnb   = (const float*)d_in[8];
  const float* cosr  = (const float*)d_in[9];
  const float* sinr  = (const float*)d_in[10];
  const float* hadq  = (const float*)d_in[11];
  const float* hadk  = (const float*)d_in[12];
  const int*   kidx  = (const int*)d_in[15];

  float* out = (float*)d_out;
  const size_t o_qs = (size_t)BS * Hn * D;    // 33,554,432
  const size_t o_kh = o_qs + (size_t)BS * Hn; // +   262,144
  const size_t o_ks = o_kh + (size_t)L * D;   // + 2,097,152
  const size_t o_w  = o_ks + (size_t)L;       // +    16,384

  // workspace layout (bf16 buffers)
  u16* xq_bf   = (u16*)d_ws;
  u16* xk_bf   = xq_bf + (size_t)BS * H;
  u16* wqbT_bf = xk_bf + (size_t)BS * H;        // [4096][2048]
  u16* wkT_bf  = wqbT_bf + (size_t)H * Hn * D;  // [128][2048]
  u16* hadq_bf = wkT_bf + (size_t)H * D;
  u16* hadk_bf = hadq_bf + (size_t)D * D;

  // seed k-cache output region with the (zero) input cache; scatter overwrites
  (void)hipMemcpyAsync(out + o_kh, d_in[13], sizeof(float) * (size_t)L * D,
                       hipMemcpyDeviceToDevice, stream);
  (void)hipMemcpyAsync(out + o_ks, d_in[14], sizeof(float) * (size_t)L,
                       hipMemcpyDeviceToDevice, stream);

  // one-time weight prep: bf16 + transpose to [N][K]
  transpose_f2bf_kernel<<<dim3((Hn * D) / 32, H / 32), 256, 0, stream>>>(
      w_qb, wqbT_bf, H, Hn * D);
  transpose_f2bf_kernel<<<dim3(D / 32, H / 32), 256, 0, stream>>>(wk, wkT_bf,
                                                                  H, D);
  f2bf_kernel<<<(D * D + 255) / 256, 256, 0, stream>>>(hadq, hadq_bf, D * D);
  f2bf_kernel<<<(D * D + 255) / 256, 256, 0, stream>>>(hadk, hadk_bf, D * D);

  prep_kernel<<<BS, 256, 0, stream>>>(x, qn, qns, lng, lnb, wproj, xq_bf,
                                      xk_bf, out + o_w);

  fused_branch_kernel<false><<<dim3(BS / 128, Hn), 256, 0, stream>>>(
      xq_bf, wqbT_bf, hadq_bf, w_qbs, cosr, sinr, nullptr, out, out + o_qs);
  fused_branch_kernel<true><<<dim3(BS / 128, 1), 256, 0, stream>>>(
      xk_bf, wkT_bf, hadk_bf, nullptr, cosr, sinr, kidx, out + o_kh,
      out + o_ks);
}
